// Attention2d_60730837565713
// MI455X (gfx1250) — matrix operand order repaired and checked by reading, not compiled, not run on hardware
//
#include <hip/hip_runtime.h>
#include <math.h>

// ---------------------------------------------------------------------------
// Attention2d (LeViT-style) for MI455X / gfx1250, wave32 + WMMA f16.
//
// Pipeline (all f16 intermediates, f32 WMMA accumulation):
//  1. convert weights f32->f16, transpose x -> xT[b][n][c] (f16)
//  2. Q/K/V 1x1-conv GEMMs  (v_wmma_f32_16x16x32_f16)
//  3. S = scale*Q.K^T + relpos bias          (WMMA, K=32 = one step)
//  4. fused th1-mix -> softmax -> th2-mix, in-register across all 8 heads
//  5. O = P.V, fused depthwise-3x3 + erf-GELU epilogue  (WMMA)
//  6. out = proj_w @ G + proj_b                          (WMMA)
// ---------------------------------------------------------------------------

typedef __attribute__((ext_vector_type(16))) _Float16 v16h;
typedef __attribute__((ext_vector_type(8)))  _Float16 v8h;
typedef __attribute__((ext_vector_type(8)))  float    v8f;

#define DEV static __device__ __forceinline__

static constexpr int Bq  = 8;
static constexpr int Cc  = 512;
static constexpr int Hh  = 32;
static constexpr int Ww  = 32;
static constexpr int NN  = 1024;   // H*W
static constexpr int NH  = 8;
static constexpr int DK  = 32;
static constexpr int KH  = 256;    // NH*DK
static constexpr int DH  = 1024;   // 4*DK*NH
static constexpr int DV  = 128;    // DH/NH

DEV v8f wmma_f16(v16h a, v16h b, v8f c) {
  // 8 args: (neg_a, A, neg_b, B, c_mod, C, reuse_a, reuse_b)
  return __builtin_amdgcn_wmma_f32_16x16x32_f16(false, a, false, b, (short)0, c,
                                                false, false);
}

DEV v8f zero8() {
  v8f z;
#pragma unroll
  for (int i = 0; i < 8; ++i) z[i] = 0.0f;
  return z;
}

// A-fragment (16x32 f16, M x K). src points at (row0, k0) of a row-major
// matrix with leading dim `ld` halfs.  ISA layout: lane L -> row L&15,
// e=0..7 -> k = 8*(L>>4)+e ; e=8..15 -> k = 16+8*(L>>4)+(e-8).
DEV v16h loadA(const _Float16* __restrict__ src, size_t ld, unsigned lane) {
  const _Float16* p = src + (size_t)(lane & 15u) * ld + ((lane >> 4) << 3);
  union { v16h v; v8h h8[2]; } u;
  u.h8[0] = *(const v8h*)(p);
  u.h8[1] = *(const v8h*)(p + 16);
  return u.v;
}

// B-fragment (32x16 f16, K x N), stored "column-contiguous": element (k, col)
// at src[col*ldc + k].  ISA layout: lane L -> col L&15, e=0..7 -> k = 8*(L>>4)+e ;
// e=8..15 -> k = 16+8*(L>>4)+(e-8).
DEV v16h loadB(const _Float16* __restrict__ src, size_t ldc, unsigned lane) {
  const _Float16* p = src + (size_t)(lane & 15u) * ldc + ((lane >> 4) << 3);
  union { v16h v; v8h h8[2]; } u;
  u.h8[0] = *(const v8h*)(p);
  u.h8[1] = *(const v8h*)(p + 16);
  return u.v;
}
// C/D fragment element r (0..7) at lane L -> (row = r + 8*(L>>4), col = L&15)

// ---------------------------------------------------------------------------
// conversion kernels
// ---------------------------------------------------------------------------
__global__ void k_cvt(const float* __restrict__ x, _Float16* __restrict__ y, int n) {
  int i = blockIdx.x * 256 + threadIdx.x;
  if (i < n) y[i] = (_Float16)x[i];
}

// x: [B, C, N] f32 -> xT: [B, N, C] f16  (coalesced writes)
__global__ void k_xT(const float* __restrict__ x, _Float16* __restrict__ xT) {
  int i = blockIdx.x * 256 + threadIdx.x;          // B*N*C total
  int b = i / (NN * Cc);
  int r = i - b * (NN * Cc);
  int n = r / Cc, c = r - n * Cc;
  xT[i] = (_Float16)x[((size_t)b * Cc + c) * NN + n];
}

// ---------------------------------------------------------------------------
// 1x1-conv GEMM:  out[o][n] = sum_c W[o][c]*xT[n][c] + bias[o]
// mode 0: store [b][h][n][dk] (q / k-transposed);  mode 1: store [b][o][n] (v)
// block 128 (4 waves), wave tile 16(o) x 128(n), grid (O/64, N/128, B)
// ---------------------------------------------------------------------------
__global__ void k_proj(const _Float16* __restrict__ Wh,   // [O, 512]
                       const _Float16* __restrict__ xT,   // [B, N, 512]
                       const float*    __restrict__ bias, // [O]
                       _Float16*       __restrict__ out,
                       int O, int mode) {
  unsigned lane = threadIdx.x & 31u, wave = threadIdx.x >> 5;
  int o0 = blockIdx.x * 64 + wave * 16;
  int n0 = blockIdx.y * 128;
  int b  = blockIdx.z;
  const _Float16* xb = xT + ((size_t)b * NN + n0) * Cc;

  v8f acc[8];
#pragma unroll
  for (int f = 0; f < 8; ++f) acc[f] = zero8();

  for (int k = 0; k < Cc; k += 32) {
    v16h a = loadA(Wh + (size_t)o0 * Cc + k, Cc, lane);
#pragma unroll
    for (int f = 0; f < 8; ++f) {
      v16h bm = loadB(xb + (size_t)(f * 16) * Cc + k, Cc, lane);
      acc[f] = wmma_f16(a, bm, acc[f]);
    }
  }

#pragma unroll
  for (int f = 0; f < 8; ++f) {
#pragma unroll
    for (int r = 0; r < 8; ++r) {
      int o = o0 + r + 8 * (int)(lane >> 4);
      int n = n0 + f * 16 + (int)(lane & 15u);
      float v = acc[f][r] + bias[o];
      if (mode == 0) {
        int h = o >> 5, dk = o & 31;
        out[(((size_t)b * NH + h) * NN + n) * DK + dk] = (_Float16)v;
      } else {
        out[((size_t)b * O + o) * NN + n] = (_Float16)v;
      }
    }
  }
}

// ---------------------------------------------------------------------------
// S[bh][n][m] = scale * (q . k) + attn_bias[h][relpos(n,m)]   (K = 32, 1 step)
// grid (N/64, N/128, B*NH), block 128
// ---------------------------------------------------------------------------
__global__ void k_scores(const _Float16* __restrict__ Q,   // [B,8,N,32]
                         const _Float16* __restrict__ Kt,  // [B,8,N,32]
                         const float*    __restrict__ abias, // [8,1024]
                         _Float16*       __restrict__ S) { // [B,8,N,N]
  unsigned lane = threadIdx.x & 31u, wave = threadIdx.x >> 5;
  int n0 = blockIdx.x * 64 + wave * 16;
  int m0 = blockIdx.y * 128;
  int bh = blockIdx.z;
  int h  = bh & 7;
  const _Float16* qb = Q  + ((size_t)bh * NN + n0) * DK;
  const _Float16* kb = Kt + ((size_t)bh * NN + m0) * DK;

  v16h a = loadA(qb, DK, lane);
  v8f acc[8];
#pragma unroll
  for (int f = 0; f < 8; ++f) {
    v16h bm = loadB(kb + (size_t)(f * 16) * DK, DK, lane);
    acc[f] = wmma_f16(a, bm, zero8());
  }

  const float scale = 0.17677669529663689f;  // KEY_DIM**-0.5
#pragma unroll
  for (int f = 0; f < 8; ++f) {
#pragma unroll
    for (int r = 0; r < 8; ++r) {
      int n = n0 + r + 8 * (int)(lane >> 4);
      int m = m0 + f * 16 + (int)(lane & 15u);
      int yn = n >> 5, xn = n & 31, ym = m >> 5, xm = m & 31;
      int dy = yn > ym ? yn - ym : ym - yn;
      int dx = xn > xm ? xn - xm : xm - xn;
      float v = acc[f][r] * scale + abias[h * NN + dy * Ww + dx];
      S[((size_t)bh * NN + n) * NN + m] = (_Float16)v;
    }
  }
}

// ---------------------------------------------------------------------------
// Fused talking-head-1 -> softmax(dim m) -> talking-head-2, in place on S.
// One block per (n, b) row; each thread holds 4 m's for ALL 8 heads.
// ---------------------------------------------------------------------------
__global__ void k_mix_softmax(_Float16* __restrict__ S,      // [B,8,N,N]
                              const float* __restrict__ w1, const float* __restrict__ bb1,
                              const float* __restrict__ w2, const float* __restrict__ bb2) {
  __shared__ float sw1[64], sw2[64], sb1[8], sb2[8];
  __shared__ float red[256];
  int t = threadIdx.x;
  if (t < 64) { sw1[t] = w1[t]; sw2[t] = w2[t]; }
  if (t < 8)  { sb1[t] = bb1[t]; sb2[t] = bb2[t]; }
  __syncthreads();

  int n = blockIdx.x, b = blockIdx.y;

  float s[8][4];
#pragma unroll
  for (int h = 0; h < 8; ++h)
#pragma unroll
    for (int j = 0; j < 4; ++j)
      s[h][j] = (float)S[(((size_t)b * NH + h) * NN + n) * NN + t + 256 * j];

  float p[8][4];
#pragma unroll
  for (int g = 0; g < 8; ++g)
#pragma unroll
    for (int j = 0; j < 4; ++j) {
      float acc = sb1[g];
#pragma unroll
      for (int h = 0; h < 8; ++h) acc += sw1[g * 8 + h] * s[h][j];
      p[g][j] = acc;
    }

  for (int g = 0; g < 8; ++g) {
    // row max
    float m = fmaxf(fmaxf(p[g][0], p[g][1]), fmaxf(p[g][2], p[g][3]));
    red[t] = m; __syncthreads();
    for (int st = 128; st > 0; st >>= 1) {
      if (t < st) red[t] = fmaxf(red[t], red[t + st]);
      __syncthreads();
    }
    float mx = red[0]; __syncthreads();
    // exp + row sum
    float sum = 0.0f;
#pragma unroll
    for (int j = 0; j < 4; ++j) { p[g][j] = __expf(p[g][j] - mx); sum += p[g][j]; }
    red[t] = sum; __syncthreads();
    for (int st = 128; st > 0; st >>= 1) {
      if (t < st) red[t] += red[t + st];
      __syncthreads();
    }
    float inv = 1.0f / red[0]; __syncthreads();
#pragma unroll
    for (int j = 0; j < 4; ++j) p[g][j] *= inv;
  }

#pragma unroll
  for (int g2 = 0; g2 < 8; ++g2)
#pragma unroll
    for (int j = 0; j < 4; ++j) {
      float acc = sb2[g2];
#pragma unroll
      for (int g = 0; g < 8; ++g) acc += sw2[g2 * 8 + g] * p[g][j];
      S[(((size_t)b * NH + g2) * NN + n) * NN + t + 256 * j] = (_Float16)acc;
    }
}

// ---------------------------------------------------------------------------
// O = P.V  (K = 1024), fused depthwise-3x3(v) + erf-GELU epilogue.
// Output Og[b][n][c] f16 (proj-GEMM B layout).  grid (N/64, NH, B), block 128.
// ---------------------------------------------------------------------------
__global__ void k_pv(const _Float16* __restrict__ S,    // [B,8,N,N] probs
                     const _Float16* __restrict__ V,    // [B,1024,N]  v[b][c][m]
                     const float*    __restrict__ vlw,  // [1024*9]
                     const float*    __restrict__ vlb,  // [1024]
                     _Float16*       __restrict__ Og) { // [B, N, 1024]
  unsigned lane = threadIdx.x & 31u, wave = threadIdx.x >> 5;
  int n0 = blockIdx.x * 64 + wave * 16;
  int h  = blockIdx.y;
  int b  = blockIdx.z;
  const _Float16* Pb = S + (((size_t)b * NH + h) * NN + n0) * NN;
  const _Float16* Vb = V + ((size_t)b * DH + h * DV) * NN;  // [d][m], ldc=N

  v8f acc[8];
#pragma unroll
  for (int f = 0; f < 8; ++f) acc[f] = zero8();

  for (int k = 0; k < NN; k += 32) {
    v16h a = loadA(Pb + k, NN, lane);
#pragma unroll
    for (int f = 0; f < 8; ++f) {
      v16h bm = loadB(Vb + (size_t)(f * 16) * NN + k, NN, lane);
      acc[f] = wmma_f16(a, bm, acc[f]);
    }
  }

#pragma unroll
  for (int f = 0; f < 8; ++f) {
#pragma unroll
    for (int r = 0; r < 8; ++r) {
      int n = n0 + r + 8 * (int)(lane >> 4);
      int d = f * 16 + (int)(lane & 15u);
      int c = h * DV + d;
      // depthwise 3x3 on v (channel c) at spatial n
      int y = n >> 5, x = n & 31;
      float vl = vlb[c];
      const _Float16* vsrc = V + ((size_t)b * DH + c) * NN;
      const float* wv = vlw + c * 9;
      for (int ky = 0; ky < 3; ++ky) {
        int yy = y + ky - 1;
        if (yy < 0 || yy >= Hh) continue;
        for (int kx = 0; kx < 3; ++kx) {
          int xx = x + kx - 1;
          if (xx < 0 || xx >= Ww) continue;
          vl += wv[ky * 3 + kx] * (float)vsrc[yy * Ww + xx];
        }
      }
      float g  = acc[f][r] + vl;
      float ge = 0.5f * g * (1.0f + erff(g * 0.70710678118654752f));
      Og[((size_t)b * NN + n) * DH + c] = (_Float16)ge;
    }
  }
}

// ---------------------------------------------------------------------------
// out[b][c][n] = proj_w @ G + proj_b   (K = 1024)
// grid (C/64, N/128, B), block 128
// ---------------------------------------------------------------------------
__global__ void k_out(const _Float16* __restrict__ Pw,  // [512, 1024] f16
                      const _Float16* __restrict__ Og,  // [B, N, 1024]
                      const float*    __restrict__ pb,  // [512]
                      float*          __restrict__ out) {
  unsigned lane = threadIdx.x & 31u, wave = threadIdx.x >> 5;
  int c0 = blockIdx.x * 64 + wave * 16;
  int n0 = blockIdx.y * 128;
  int b  = blockIdx.z;
  const _Float16* Bb = Og + ((size_t)b * NN + n0) * DH;

  v8f acc[8];
#pragma unroll
  for (int f = 0; f < 8; ++f) acc[f] = zero8();

  for (int k = 0; k < DH; k += 32) {
    v16h a = loadA(Pw + (size_t)c0 * DH + k, DH, lane);
#pragma unroll
    for (int f = 0; f < 8; ++f) {
      v16h bm = loadB(Bb + (size_t)(f * 16) * DH + k, DH, lane);
      acc[f] = wmma_f16(a, bm, acc[f]);
    }
  }

#pragma unroll
  for (int f = 0; f < 8; ++f) {
#pragma unroll
    for (int r = 0; r < 8; ++r) {
      int c = c0 + r + 8 * (int)(lane >> 4);
      int n = n0 + f * 16 + (int)(lane & 15u);
      out[((size_t)b * Cc + c) * NN + n] = acc[f][r] + pb[c];
    }
  }
}

// ---------------------------------------------------------------------------
// host launcher
// ---------------------------------------------------------------------------
static inline size_t align256(size_t x) { return (x + 255) & ~(size_t)255; }

extern "C" void kernel_launch(void* const* d_in, const int* in_sizes, int n_in,
                              void* d_out, int out_size, void* d_ws, size_t ws_size,
                              hipStream_t stream) {
  const float* x      = (const float*)d_in[0];
  const float* q_w    = (const float*)d_in[1];
  const float* q_b    = (const float*)d_in[2];
  const float* k_w    = (const float*)d_in[3];
  const float* k_b    = (const float*)d_in[4];
  const float* v_w    = (const float*)d_in[5];
  const float* v_b    = (const float*)d_in[6];
  const float* vl_w   = (const float*)d_in[7];
  const float* vl_b   = (const float*)d_in[8];
  const float* th1_w  = (const float*)d_in[9];
  const float* th1_b  = (const float*)d_in[10];
  const float* th2_w  = (const float*)d_in[11];
  const float* th2_b  = (const float*)d_in[12];
  const float* proj_w = (const float*)d_in[13];
  const float* proj_b = (const float*)d_in[14];
  const float* abias  = (const float*)d_in[15];
  float* out = (float*)d_out;

  // workspace bump allocation (f16 elements), total ~179 MB
  char* base = (char*)d_ws;
  size_t off = 0;
  auto alloc_h = [&](size_t nelem) -> _Float16* {
    _Float16* p = (_Float16*)(base + off);
    off = align256(off + nelem * sizeof(_Float16));
    return p;
  };
  _Float16* xT  = alloc_h((size_t)Bq * NN * Cc);       // 8 MB
  _Float16* qwh = alloc_h((size_t)KH * Cc);
  _Float16* kwh = alloc_h((size_t)KH * Cc);
  _Float16* vwh = alloc_h((size_t)DH * Cc);
  _Float16* pwh = alloc_h((size_t)Cc * DH);
  _Float16* Qw  = alloc_h((size_t)Bq * NH * NN * DK);  // 4 MB
  _Float16* Kw  = alloc_h((size_t)Bq * NH * NN * DK);  // 4 MB
  _Float16* Vw  = alloc_h((size_t)Bq * DH * NN);       // 16 MB
  _Float16* Sw  = alloc_h((size_t)Bq * NH * NN * NN);  // 128 MB
  _Float16* Ogw = alloc_h((size_t)Bq * NN * DH);       // 16 MB
  (void)ws_size; (void)n_in; (void)in_sizes; (void)out_size;

  // 1. conversions
  k_cvt<<<(KH * Cc + 255) / 256, 256, 0, stream>>>(q_w, qwh, KH * Cc);
  k_cvt<<<(KH * Cc + 255) / 256, 256, 0, stream>>>(k_w, kwh, KH * Cc);
  k_cvt<<<(DH * Cc + 255) / 256, 256, 0, stream>>>(v_w, vwh, DH * Cc);
  k_cvt<<<(Cc * DH + 255) / 256, 256, 0, stream>>>(proj_w, pwh, Cc * DH);
  k_xT<<<(Bq * NN * Cc) / 256, 256, 0, stream>>>(x, xT);

  // 2. Q/K/V projections
  k_proj<<<dim3(KH / 64, NN / 128, Bq), 128, 0, stream>>>(qwh, xT, q_b, Qw, KH, 0);
  k_proj<<<dim3(KH / 64, NN / 128, Bq), 128, 0, stream>>>(kwh, xT, k_b, Kw, KH, 0);
  k_proj<<<dim3(DH / 64, NN / 128, Bq), 128, 0, stream>>>(vwh, xT, v_b, Vw, DH, 1);

  // 3. scores + relpos bias
  k_scores<<<dim3(NN / 64, NN / 128, Bq * NH), 128, 0, stream>>>(Qw, Kw, abias, Sw);

  // 4. th1 -> softmax -> th2 (in place)
  k_mix_softmax<<<dim3(NN, Bq), 256, 0, stream>>>(Sw, th1_w, th1_b, th2_w, th2_b);

  // 5. P.V + depthwise + GELU
  k_pv<<<dim3(NN / 64, NH, Bq), 128, 0, stream>>>(Sw, Vw, vl_w, vl_b, Ogw);

  // 6. output projection
  k_out<<<dim3(Cc / 64, NN / 128, Bq), 128, 0, stream>>>(pwh, Ogw, proj_b, out);
}
